// DualSSM_34737695490281
// MI455X (gfx1250) — compile-verified, matrix operand order checked by reading, not run on hardware
//
#include <hip/hip_runtime.h>
#include <hip/hip_bf16.h>

// ---------------------------------------------------------------------------
// DualSSM for MI455X (gfx1250, wave32, WMMA f16 16x16x32)
//
// B=256, W=512, FEAT=128, TIME=32, OV=32, IN=192, HG=256, HL=128, OG=128, OL=64
//
// Pipeline:
//   0) cvt_kernel          : fp32 weights -> f16 copies in workspace
//   1) encoder_kernel      : overlap MLP + concat -> inp f16, time-major (W,B,192)
//   2) xih_kernel          : inp @ [Wih_g;Wih_l]^T + bih  (WMMA), tiled f16 layout
//   3) scan_kernel         : persistent dual-GRU scan, Whh in registers as
//                            WMMA B-fragments, h in LDS in A-fragment layout,
//                            fused output projections (pipelined)
// ---------------------------------------------------------------------------

typedef __attribute__((ext_vector_type(16))) _Float16 v16h;
typedef __attribute__((ext_vector_type(8)))  _Float16 v8h;
typedef __attribute__((ext_vector_type(8)))  float    v8f;

#define WMMA_F16(a, b, c) \
  __builtin_amdgcn_wmma_f32_16x16x32_f16(false, (a), false, (b), (short)0, (c), false, false)

// A/B fragment gather from row-major f16 (stride in elements, multiple of 8).
// Layout (ISA 7.12.2, 16-bit A 16x32): lane L: row = L&15, kbase = (L>>4)*8,
// v0..v3 hold k = kbase..kbase+7, v4..v7 hold k = 16+kbase..16+kbase+7.
// B (32x16) mirrors with lane -> column of B (= row of W).
__device__ __forceinline__ v16h frag_from_global(const _Float16* __restrict__ base, int stride) {
  const int lane = threadIdx.x & 31;
  const int r    = lane & 15;
  const int kb   = (lane >> 4) << 3;
  const _Float16* p = base + (size_t)r * stride + kb;
  v8h lo = *(const v8h*)(p);
  v8h hi = *(const v8h*)(p + 16);
  return __builtin_shufflevector(lo, hi, 0, 1, 2, 3, 4, 5, 6, 7, 8, 9, 10, 11, 12, 13, 14, 15);
}

// Fragment load from LDS staged in per-lane layout: 16 halves (32B) per lane.
__device__ __forceinline__ v16h frag_from_lds(const _Float16* p512) {
  const int lane = threadIdx.x & 31;
  const _Float16* p = p512 + lane * 16;
  v8h lo = *(const v8h*)(p);
  v8h hi = *(const v8h*)(p + 8);
  return __builtin_shufflevector(lo, hi, 0, 1, 2, 3, 4, 5, 6, 7, 8, 9, 10, 11, 12, 13, 14, 15);
}

// ---------------------------------------------------------------------------
// 0) fp32 -> f16 weight conversion
// ---------------------------------------------------------------------------
__global__ void cvt_kernel(const float* __restrict__ src, _Float16* __restrict__ dst, int n) {
  int i = blockIdx.x * 256 + threadIdx.x;
  if (i < n) dst[i] = (_Float16)src[i];
}

// ---------------------------------------------------------------------------
// 1) overlap encoder + concat; writes inp f16 time-major: row = w*256 + b
// ---------------------------------------------------------------------------
__global__ __launch_bounds__(64) void encoder_kernel(
    const float* __restrict__ f_seq, const float* __restrict__ delta_seq,
    const float* __restrict__ ov_seq,
    const float* __restrict__ W1, const float* __restrict__ b1,
    const float* __restrict__ W2, const float* __restrict__ b2,
    _Float16* __restrict__ inp) {
  const int row = blockIdx.x;     // row = w*256 + b
  const int w = row >> 8;
  const int b = row & 255;
  const int t = threadIdx.x;      // 64 threads

  __shared__ float ovin[32];
  __shared__ float hid[64];

  const float* ovp = ov_seq + ((size_t)b * 512 + w) * 32;
  if (t < 32) ovin[t] = ovp[t];
  __syncthreads();

  float acc = b1[t];
#pragma unroll
  for (int k = 0; k < 32; ++k) acc += W1[t * 32 + k] * ovin[k];
  hid[t] = acc / (1.f + __expf(-acc));  // SiLU
  __syncthreads();

  _Float16* out = inp + (size_t)row * 192;
  const float* fp = f_seq + ((size_t)b * 512 + w) * 128;
  out[t]      = (_Float16)fp[t];
  out[64 + t] = (_Float16)fp[64 + t];
  const float* dp = delta_seq + ((size_t)b * 512 + w) * 32;
  if (t < 32) {
    out[128 + t] = (_Float16)dp[t];
  } else {
    const int o = t - 32;
    float a = b2[o];
#pragma unroll
    for (int k = 0; k < 64; ++k) a += W2[o * 64 + k] * hid[k];
    out[160 + o] = (_Float16)a;
  }
}

// ---------------------------------------------------------------------------
// 2) xih = inp @ WihC^T + bih.  M=131072, N=1152, K=192.
//    Output layout: f16 tiles [mtile (8192)][ntile (72)][n (16)][m (16)].
//    Grid: 1024 blocks x 256 threads (8 waves). Wave = one M-tile; loops over
//    18 column groups of 64, B staged in LDS in fragment layout.
// ---------------------------------------------------------------------------
__global__ __launch_bounds__(256) void xih_kernel(
    const _Float16* __restrict__ inp, const _Float16* __restrict__ WihC,
    const float* __restrict__ bih_g, const float* __restrict__ bih_l,
    _Float16* __restrict__ xih) {
  __shared__ __align__(32) _Float16 bsm[4 * 6 * 512];  // 24 KB
  const int wave = threadIdx.x >> 5;
  const int lane = threadIdx.x & 31;
  const int mtile = blockIdx.x * 8 + wave;
  const int nn = lane & 15;
  const int mb = (lane >> 4) << 3;

  const _Float16* arow = inp + (size_t)mtile * 16 * 192;
  v16h afrag[6];
#pragma unroll
  for (int kf = 0; kf < 6; ++kf) afrag[kf] = frag_from_global(arow + kf * 32, 192);

  for (int cg = 0; cg < 18; ++cg) {
    __syncthreads();
    // stage 64x192 B chunk as fragments: 4 ntiles x 6 kfrags x 32 lanes
    for (int i = threadIdx.x; i < 768; i += 256) {
      const int l  = i & 31;
      const int kf = (i >> 5) % 6;
      const int nt = i / 192;
      const int n  = cg * 64 + nt * 16 + (l & 15);
      const int kb = ((l >> 4) << 3) + kf * 32;
      const _Float16* p = WihC + (size_t)n * 192 + kb;
      v8h lo = *(const v8h*)(p);
      v8h hi = *(const v8h*)(p + 16);
      _Float16* d = bsm + (nt * 6 + kf) * 512 + l * 16;
      *(v8h*)(d)     = lo;
      *(v8h*)(d + 8) = hi;
    }
    __syncthreads();

#pragma unroll
    for (int nt = 0; nt < 4; ++nt) {
      v8f c = {0.f, 0.f, 0.f, 0.f, 0.f, 0.f, 0.f, 0.f};
#pragma unroll
      for (int kf = 0; kf < 6; ++kf) {
        v16h bf = frag_from_lds(bsm + (nt * 6 + kf) * 512);
        c = WMMA_F16(afrag[kf], bf, c);
      }
      const int col = cg * 64 + nt * 16 + nn;
      const float bias = (col < 768) ? bih_g[col] : bih_l[col - 768];
      const int ntg = cg * 4 + nt;
      _Float16* d = xih + ((size_t)mtile * 72 + ntg) * 256 + nn * 16 + mb;
      v8h o;
#pragma unroll
      for (int j = 0; j < 8; ++j) o[j] = (_Float16)(c[j] + bias);
      *(v8h*)d = o;
    }
  }
}

// ---------------------------------------------------------------------------
// 3) persistent GRU scan.  Blocks 0..15: global GRU (H=256,OUT=128);
//    blocks 16..31: local GRU (H=128,OUT=64).  1024 threads = 32 waves.
//    Waves 0..23: recurrent GEMM (Whh B-frags in registers, loop-invariant).
//    Waves 24..24+OUT/16-1: output projection (pipelined one step behind).
//    h lives in LDS in A-fragment layout (2x ds_load_b128 per fragment).
// ---------------------------------------------------------------------------
template <int H, int OUT>
__device__ void gru_scan_impl(const _Float16* __restrict__ xih, int ntoff,
                              const _Float16* __restrict__ Whh,
                              const _Float16* __restrict__ Wproj,
                              const float* __restrict__ bhh,
                              const float* __restrict__ bproj,
                              float* __restrict__ out,
                              _Float16* hA, float* gh, int wg) {
  constexpr int NKF = H / 32;        // k-fragments per row-block (8 / 4)
  constexpr int NT  = (3 * H) / 16;  // recurrent N tiles (48 / 24)
  constexpr int TPW = NT / 24;       // N tiles per recurrent wave (2 / 1)
  constexpr int PW  = OUT / 16;      // projection waves (8 / 4)

  const int wave = threadIdx.x >> 5;
  const int lane = threadIdx.x & 31;
  const bool isRec  = wave < 24;
  const bool isProj = (wave >= 24) && (wave < 24 + PW);
  const int nn = lane & 15;
  const int mb = (lane >> 4) << 3;

  // Loop-invariant weight fragments in VGPRs. Projection waves reuse bw[0][*].
  v16h bw[TPW][NKF];
  if (isRec) {
#pragma unroll
    for (int tpw = 0; tpw < TPW; ++tpw) {
      const int nt = wave * TPW + tpw;
#pragma unroll
      for (int kf = 0; kf < NKF; ++kf)
        bw[tpw][kf] = frag_from_global(Whh + (size_t)nt * 16 * H + kf * 32, H);
    }
  } else if (isProj) {
    const int nt = wave - 24;
#pragma unroll
    for (int kf = 0; kf < NKF; ++kf)
      bw[0][kf] = frag_from_global(Wproj + (size_t)nt * 16 * H + kf * 32, H);
  }

  // h0 = 0
  for (int i = threadIdx.x; i < NKF * 512; i += 1024) hA[i] = (_Float16)0.f;
  __syncthreads();

  auto proj_store = [&](int tt) {
    v8f c = {0.f, 0.f, 0.f, 0.f, 0.f, 0.f, 0.f, 0.f};
#pragma unroll
    for (int kf = 0; kf < NKF; ++kf) {
      v16h a = frag_from_lds(hA + kf * 512);
      c = WMMA_F16(a, bw[0][kf], c);
    }
    const int n = (wave - 24) * 16 + nn;
    const float bias = bproj[n];
#pragma unroll
    for (int j = 0; j < 8; ++j)
      out[((size_t)(wg * 16 + mb + j) * 512 + tt) * OUT + n] = c[j] + bias;
  };

  for (int t = 0; t < 512; ++t) {
    // ---- phase 1: gh = h @ Whh^T (rec waves); out[t-1] (proj waves) ----
    if (isRec) {
#pragma unroll
      for (int tpw = 0; tpw < TPW; ++tpw) {
        v8f c = {0.f, 0.f, 0.f, 0.f, 0.f, 0.f, 0.f, 0.f};
#pragma unroll
        for (int kf = 0; kf < NKF; ++kf) {
          v16h a = frag_from_lds(hA + kf * 512);
          c = WMMA_F16(a, bw[tpw][kf], c);
        }
        const int nt = wave * TPW + tpw;
        *(v8f*)(gh + nt * 256 + nn * 16 + mb) = c;  // D-fragment tile layout
      }
    } else if (isProj) {
      if (t > 0) proj_store(t - 1);
    }
    __syncthreads();

    // ---- phase 2: gates + state update ----
    const size_t xrow = (size_t)(t * 16 + wg) * 72 * 256;
    for (int i = threadIdx.x; i < 16 * H; i += 1024) {
      const int m = i / H;
      const int u = i - m * H;
      const int c16 = (u & 15) * 16 + m;
      const float hr = gh[(u >> 4) * 256 + c16] + bhh[u];
      const float hz = gh[((H + u) >> 4) * 256 + c16] + bhh[H + u];
      const float hn = gh[((2 * H + u) >> 4) * 256 + c16] + bhh[2 * H + u];
      const float xr = (float)xih[xrow + (size_t)(ntoff + (u >> 4)) * 256 + c16];
      const float xz = (float)xih[xrow + (size_t)(ntoff + ((H + u) >> 4)) * 256 + c16];
      const float xn = (float)xih[xrow + (size_t)(ntoff + ((2 * H + u) >> 4)) * 256 + c16];
      const float r = 1.f / (1.f + __expf(-(xr + hr)));
      const float z = 1.f / (1.f + __expf(-(xz + hz)));
      const float ng = tanhf(xn + r * hn);
      // h element (m,u) inside A-fragment LDS layout
      const int kf = u >> 5, kl = u & 31, sub = kl & 7, grp = kl >> 3;
      const int hidx = kf * 512 + (((grp & 1) << 4) + m) * 16 + ((grp >> 1) << 3) + sub;
      const float hp = (float)hA[hidx];
      hA[hidx] = (_Float16)((1.f - z) * ng + z * hp);
    }
    __syncthreads();
  }
  if (isProj) proj_store(511);  // drain pipelined projection
}

__global__ __launch_bounds__(1024, 1) void scan_kernel(
    const _Float16* __restrict__ xih,
    const _Float16* __restrict__ WhhG, const _Float16* __restrict__ WhhL,
    const _Float16* __restrict__ WgH,  const _Float16* __restrict__ WlH,
    const float* __restrict__ bhh_g, const float* __restrict__ bhh_l,
    const float* __restrict__ bg, const float* __restrict__ bl,
    float* __restrict__ g_out, float* __restrict__ m_out) {
  __shared__ __align__(32) _Float16 hA[8 * 512];   // h in A-fragment layout (8 KB)
  __shared__ __align__(32) float    gh[16 * 768];  // recurrent GEMM scratch (48 KB)
  if (blockIdx.x < 16)
    gru_scan_impl<256, 128>(xih, 0, WhhG, WgH, bhh_g, bg, g_out, hA, gh, blockIdx.x);
  else
    gru_scan_impl<128, 64>(xih, 48, WhhL, WlH, bhh_l, bl, m_out, hA, gh, blockIdx.x - 16);
}

// ---------------------------------------------------------------------------
// launch
// ---------------------------------------------------------------------------
extern "C" void kernel_launch(void* const* d_in, const int* in_sizes, int n_in,
                              void* d_out, int out_size, void* d_ws, size_t ws_size,
                              hipStream_t stream) {
  const float* f_seq = (const float*)d_in[0];
  const float* delta = (const float*)d_in[1];
  const float* ovs   = (const float*)d_in[2];
  const float* W1    = (const float*)d_in[3];
  const float* b1    = (const float*)d_in[4];
  const float* W2    = (const float*)d_in[5];
  const float* b2    = (const float*)d_in[6];
  const float* Wih_g = (const float*)d_in[7];
  const float* Whh_g = (const float*)d_in[8];
  const float* bih_g = (const float*)d_in[9];
  const float* bhh_g = (const float*)d_in[10];
  const float* Wih_l = (const float*)d_in[11];
  const float* Whh_l = (const float*)d_in[12];
  const float* bih_l = (const float*)d_in[13];
  const float* bhh_l = (const float*)d_in[14];
  const float* Wg    = (const float*)d_in[15];
  const float* bg    = (const float*)d_in[16];
  const float* Wl    = (const float*)d_in[17];
  const float* bl    = (const float*)d_in[18];

  // workspace layout (bytes)
  char* ws = (char*)d_ws;
  _Float16* inp   = (_Float16*)(ws + 0ull);            // 131072*192*2   = 50,331,648
  _Float16* xih   = (_Float16*)(ws + 50331648ull);     // 131072*1152*2  = 301,989,888
  _Float16* WihC  = (_Float16*)(ws + 352321536ull);    // 1152*192*2     = 442,368
  _Float16* WhhGh = (_Float16*)(ws + 352763904ull);    // 768*256*2      = 393,216
  _Float16* WhhLh = (_Float16*)(ws + 353157120ull);    // 384*128*2      = 98,304
  _Float16* WgH   = (_Float16*)(ws + 353255424ull);    // 128*256*2      = 65,536
  _Float16* WlH   = (_Float16*)(ws + 353320960ull);    // 64*128*2       = 16,384

  cvt_kernel<<<(147456 + 255) / 256, 256, 0, stream>>>(Wih_g, WihC, 147456);
  cvt_kernel<<<(73728 + 255) / 256, 256, 0, stream>>>(Wih_l, WihC + 147456, 73728);
  cvt_kernel<<<(196608 + 255) / 256, 256, 0, stream>>>(Whh_g, WhhGh, 196608);
  cvt_kernel<<<(49152 + 255) / 256, 256, 0, stream>>>(Whh_l, WhhLh, 49152);
  cvt_kernel<<<(32768 + 255) / 256, 256, 0, stream>>>(Wg, WgH, 32768);
  cvt_kernel<<<(8192 + 255) / 256, 256, 0, stream>>>(Wl, WlH, 8192);

  encoder_kernel<<<131072, 64, 0, stream>>>(f_seq, delta, ovs, W1, b1, W2, b2, inp);
  xih_kernel<<<1024, 256, 0, stream>>>(inp, WihC, bih_g, bih_l, xih);

  float* g_out = (float*)d_out;                 // (B,W,128) fp32
  float* m_out = g_out + 256ull * 512ull * 128ull;  // (B,W,64) fp32
  scan_kernel<<<32, 1024, 0, stream>>>(xih, WhhGh, WhhLh, WgH, WlH,
                                       bhh_g, bhh_l, bg, bl, g_out, m_out);
}
